// HGNN_GCN_EDGE_WO_SH_1778116460938
// MI455X (gfx1250) — matrix operand order repaired and checked by reading, not compiled, not run on hardware
//
#include <hip/hip_runtime.h>
#include <hip/hip_bf16.h>
#include <math.h>

// ---------------------------------------------------------------------------
// HGNN GCN encoder/processor/decoder with attention, for MI455X (gfx1250).
// Dense per-node math uses v_wmma_f32_16x16x32_f16 (f16 in, f32 accum).
// Weights staged once per block into LDS as f16, transposed [n][k] so B
// fragments are aligned contiguous ds_load_b128 pairs (no per-wave re-convert).
// ---------------------------------------------------------------------------

#define N_F   100000
#define N_H   49152
#define NEDGE 400000
#define T_IN  12
#define T_OUT 12
#define HDIM  64
#define A_H   128
#define LSTR  80   // LDS row stride in halfs (160B = 32B-aligned fragments)

typedef __attribute__((ext_vector_type(16))) _Float16 v16h;
typedef __attribute__((ext_vector_type(8)))  _Float16 v8h;
typedef __attribute__((ext_vector_type(8)))  float    v8f;

__device__ static inline v8f wmma_f16(v16h a, v16h b, v8f c) {
  // (neg_a, A, neg_b, B, c_mod, C, reuse_a, reuse_b)
  return __builtin_amdgcn_wmma_f32_16x16x32_f16(
      false, a, false, b, (short)0, c, false, false);
}

// B fragment from LDS-staged transposed weights S[n*LSTR + k] (f16):
// halfs 0-7 -> K = 32*kc + 8*hs + h, halfs 8-15 -> K = 32*kc + 16 + 8*hs + h.
__device__ static inline v16h load_bfrag_lds(const _Float16* S, int jn, int kc,
                                             int mloc, int half_sel) {
  int n  = 16 * jn + mloc;
  int kb = 32 * kc + 8 * half_sel;
  v8h lo = *(const v8h*)&S[n * LSTR + kb];
  v8h hi = *(const v8h*)&S[n * LSTR + kb + 16];
  return __builtin_shufflevector(lo, hi, 0, 1, 2, 3, 4, 5, 6, 7,
                                 8, 9, 10, 11, 12, 13, 14, 15);
}

// A fragment from an LDS row (f16): halfs 0-7 -> K = 32*kc + 8*hs + h,
// halfs 8-15 -> K = 32*kc + 16 + 8*hs + h. Two aligned 16B chunks.
__device__ static inline v16h load_afrag_lds(const _Float16* Srow, int kc,
                                             int half_sel) {
  v8h lo = *(const v8h*)&Srow[32 * kc + 8 * half_sel];
  v8h hi = *(const v8h*)&Srow[32 * kc + 16 + 8 * half_sel];
  return __builtin_shufflevector(lo, hi, 0, 1, 2, 3, 4, 5, 6, 7,
                                 8, 9, 10, 11, 12, 13, 14, 15);
}

__device__ static inline float lrelu(float x) { return x > 0.f ? x : 0.01f * x; }

// ---------------------------------------------------------------------------
__global__ void zero_kernel(float* __restrict__ p, int n) {
  int i = blockIdx.x * blockDim.x + threadIdx.x;
  if (i < n) p[i] = 0.f;
}

__global__ void deg_kernel(const int* __restrict__ src, const int* __restrict__ dst,
                           float* __restrict__ dsrc, float* __restrict__ ddst, int e) {
  int i = blockIdx.x * blockDim.x + threadIdx.x;
  if (i < e) {
    atomicAdd(&dsrc[src[i]], 1.f);
    atomicAdd(&ddst[dst[i]], 1.f);
  }
}

__global__ void invsqrt_kernel(float* __restrict__ d, int n) {
  int i = blockIdx.x * blockDim.x + threadIdx.x;
  if (i < n) {
    float v = d[i];
    v = v < 1.f ? 1.f : v;
    d[i] = rsqrtf(v);
  }
}

// agg[dst, t] += nan_to_num(feat[src, t]) * c_se[src]
__global__ void enc_agg_kernel(const float* __restrict__ feat,
                               const int* __restrict__ esrc, const int* __restrict__ edst,
                               const float* __restrict__ c_se,
                               float* __restrict__ agg, int e) {
  int i = blockIdx.x * blockDim.x + threadIdx.x;
  int edge = i / T_IN, t = i - edge * T_IN;
  if (edge < e) {
    int s = esrc[edge], d = edst[edge];
    float x = feat[s * T_IN + t];
    if (!isfinite(x)) x = isnan(x) ? 0.f : copysignf(3.4028235e38f, x);
    atomicAdd(&agg[d * T_IN + t], x * c_se[s]);
  }
}

__global__ void y0_kernel(const float* __restrict__ feat, float* __restrict__ y0, int n) {
  int i = blockIdx.x * blockDim.x + threadIdx.x;
  if (i < n) {
    float x = feat[i * T_IN + (T_IN - 1)];
    if (!isfinite(x)) x = isnan(x) ? 0.f : copysignf(3.4028235e38f, x);
    y0[i] = x;
  }
}

// m[n,h] = relu( sum_t lrelu(agg[n,t]*c_de[n]*W_enc[t,h]+b_enc[t,h]) * W_proc[t,o]
//               + b_proc[o] ) * c_sd[n]
__global__ void compute_m_kernel(const float* __restrict__ agg,
                                 const float* __restrict__ c_de,
                                 const float* __restrict__ c_sd,
                                 const float* __restrict__ W_enc,  // 12x64
                                 const float* __restrict__ b_enc,  // 12x64
                                 const float* __restrict__ W_proc, // 12x12
                                 const float* __restrict__ b_proc, // 12
                                 float* __restrict__ m, int o, int n_h) {
  int i = blockIdx.x * blockDim.x + threadIdx.x;
  int n = i >> 6, h = i & 63;
  if (n < n_h) {
    float cd  = c_de[n];
    float acc = 0.f;
#pragma unroll
    for (int t = 0; t < T_IN; ++t) {
      float he = agg[n * T_IN + t] * cd * W_enc[t * HDIM + h] + b_enc[t * HDIM + h];
      acc += lrelu(he) * W_proc[t * T_OUT + o];
    }
    acc += b_proc[o];
    acc = acc > 0.f ? acc : 0.f;
    m[(size_t)n * HDIM + h] = acc * c_sd[n];
  }
}

// a[dst, h] += m[src, h]  (16 chunks of 4 floats per edge)
__global__ void dec_agg_kernel(const float* __restrict__ m,
                               const int* __restrict__ dsrc, const int* __restrict__ ddst,
                               float* __restrict__ a_buf, int e) {
  int i = blockIdx.x * blockDim.x + threadIdx.x;
  int edge = i >> 4, c = i & 15;
  if (edge < e) {
    int s = dsrc[edge], d = ddst[edge];
    const float* ms = m + (size_t)s * HDIM + c * 4;
    float*       ad = a_buf + (size_t)d * HDIM + c * 4;
    atomicAdd(&ad[0], ms[0]);
    atomicAdd(&ad[1], ms[1]);
    atomicAdd(&ad[2], ms[2]);
    atomicAdd(&ad[3], ms[3]);
  }
}

// ---------------------------------------------------------------------------
// Fused per-step decoder: space1 GEMM + attention GEMMs + softmax + output.
// 128 threads = 4 waves; each wave owns a 16-row node tile (64 rows/block).
// ---------------------------------------------------------------------------
__launch_bounds__(128)
__global__ void decoder_step_kernel(const float* __restrict__ a_buf,  // N_F x 64
                                    const float* __restrict__ c_dd,   // N_F
                                    const float* __restrict__ y_prev, // N_F
                                    const float* __restrict__ W_t,    // 64x64
                                    const float* __restrict__ b_t,    // 64
                                    const float* __restrict__ W_lin,  // 64
                                    const float* __restrict__ b_lin,  // 64
                                    const float* __restrict__ W_a1,   // 64x128
                                    const float* __restrict__ b_a1,   // 128
                                    const float* __restrict__ W_a2,   // 128
                                    const float* __restrict__ b_a2,   // 1
                                    const float* __restrict__ W_o,    // 64
                                    const float* __restrict__ b_o,    // 1
                                    float* __restrict__ y_out,        // N_F
                                    int n_f) {
  __shared__ alignas(32) _Float16 s_wt [HDIM * LSTR];  // W_t^T  [n][k], f16
  __shared__ alignas(32) _Float16 s_wa1[A_H  * LSTR];  // W_a1^T [n][k], f16
  __shared__ alignas(32) _Float16 s_sp1[64 * LSTR];    // space1 tile [row][k]
  __shared__ alignas(32) _Float16 s_sp2[64 * LSTR];    // space2 tile
  __shared__ float s_sc[2 * 64];                       // per-row attn scores

  const int tid        = threadIdx.x;
  const int wave       = tid >> 5;   // 0..3
  const int lane       = tid & 31;
  const int half_sel   = lane >> 4;  // 0: lanes 0-15, 1: lanes 16-31
  const int mloc       = lane & 15;
  const int block_row0 = blockIdx.x * 64;
  const int wave_row0  = block_row0 + wave * 16;

  // ---- Stage weights into LDS (f16, transposed [n][k]), once per block.
  for (int idx = tid; idx < HDIM * HDIM; idx += 128) {  // W_t: k-major in global
    int k = idx >> 6, n = idx & 63;
    s_wt[n * LSTR + k] = (_Float16)W_t[idx];
  }
  for (int idx = tid; idx < HDIM * A_H; idx += 128) {   // W_a1: 64x128
    int k = idx >> 7, n = idx & 127;
    s_wa1[n * LSTR + k] = (_Float16)W_a1[idx];
  }

  // ---- A fragments for GEMM1: rows of (a * c_dd), K = 0..63 in two chunks.
  v16h afrag[2];
  {
    int r  = wave_row0 + mloc;
    int rc = r < n_f ? r : n_f - 1;
    float scale = (r < n_f) ? c_dd[rc] : 0.f;
    const float* arow = a_buf + (size_t)rc * HDIM;
#pragma unroll
    for (int kc = 0; kc < 2; ++kc) {
#pragma unroll
      for (int h = 0; h < 8; ++h) {
        afrag[kc][h]     = (_Float16)(arow[32 * kc + 8 * half_sel + h] * scale);
        afrag[kc][h + 8] = (_Float16)(arow[32 * kc + 16 + 8 * half_sel + h] * scale);
      }
    }
  }

  // ---- space2 = y_prev * W_lin + b_lin (rank-1): each wave fills its rows.
  for (int idx = lane; idx < 16 * HDIM; idx += 32) {
    int rr = idx >> 6, n = idx & 63;
    int grow = wave_row0 + rr;
    float yp = grow < n_f ? y_prev[grow] : 0.f;
    s_sp2[(wave * 16 + rr) * LSTR + n] = (_Float16)(yp * W_lin[n] + b_lin[n]);
  }
  __syncthreads();  // weights staged, s_sp2 rows visible

  // ---- GEMM1: space1 = lrelu(A @ W_t + b_t) -> LDS f16 (4 N-tiles of 16)
#pragma unroll
  for (int jn = 0; jn < 4; ++jn) {
    v8f c = {};
    c = wmma_f16(afrag[0], load_bfrag_lds(s_wt, jn, 0, mloc, half_sel), c);
    c = wmma_f16(afrag[1], load_bfrag_lds(s_wt, jn, 1, mloc, half_sel), c);
    int n   = 16 * jn + mloc;
    float bt = b_t[n];
#pragma unroll
    for (int r = 0; r < 8; ++r) {
      int mm = r + 8 * half_sel;  // local row in wave tile
      s_sp1[(wave * 16 + mm) * LSTR + n] = (_Float16)lrelu(c[r] + bt);
    }
  }
  __syncthreads();

  // ---- Attention: score_br = sum_N tanh(space_br @ W_a1 + b_a1)[N]*W_a2[N] + b_a2
  const float ba2 = b_a2[0];
  for (int br = 0; br < 2; ++br) {
    const _Float16* Srow = ((br == 0) ? s_sp1 : s_sp2) + (wave * 16 + mloc) * LSTR;
    v16h af0 = load_afrag_lds(Srow, 0, half_sel);
    v16h af1 = load_afrag_lds(Srow, 1, half_sel);
    float part[8];
#pragma unroll
    for (int r = 0; r < 8; ++r) part[r] = 0.f;
#pragma unroll
    for (int jn = 0; jn < 8; ++jn) {
      v8f c = {};
      c = wmma_f16(af0, load_bfrag_lds(s_wa1, jn, 0, mloc, half_sel), c);
      c = wmma_f16(af1, load_bfrag_lds(s_wa1, jn, 1, mloc, half_sel), c);
      int n = 16 * jn + mloc;
      float ba = b_a1[n], wa2 = W_a2[n];
#pragma unroll
      for (int r = 0; r < 8; ++r) part[r] += tanhf(c[r] + ba) * wa2;
    }
    // butterfly reduce over the 16-lane N-group (stays within half-wave)
#pragma unroll
    for (int r = 0; r < 8; ++r) {
      float v = part[r];
      v += __shfl_xor(v, 1, 32);
      v += __shfl_xor(v, 2, 32);
      v += __shfl_xor(v, 4, 32);
      v += __shfl_xor(v, 8, 32);
      part[r] = v;
    }
    if (mloc == 0) {  // lane 0 holds rows 0-7, lane 16 rows 8-15
#pragma unroll
      for (int r = 0; r < 8; ++r)
        s_sc[br * 64 + wave * 16 + r + 8 * half_sel] = part[r] + ba2;
    }
  }
  __syncthreads();

  // ---- softmax over the two branches + blend + output dot
  if (tid < 64) {
    int grow = block_row0 + tid;
    if (grow < n_f) {
      float s1 = s_sc[tid], s2 = s_sc[64 + tid];
      float mx = fmaxf(s1, s2);
      float e1 = __expf(s1 - mx), e2 = __expf(s2 - mx);
      float inv = 1.f / (e1 + e2);
      float a1 = e1 * inv, a2 = e2 * inv;
      float y = b_o[0];
#pragma unroll
      for (int h = 0; h < HDIM; ++h) {
        float st = a1 * (float)s_sp1[tid * LSTR + h] + a2 * (float)s_sp2[tid * LSTR + h];
        y += st * W_o[h];
      }
      y_out[grow] = y;
    }
  }
}

// ---------------------------------------------------------------------------
static inline int cdiv(long long a, int b) { return (int)((a + b - 1) / b); }

extern "C" void kernel_launch(void* const* d_in, const int* in_sizes, int n_in,
                              void* d_out, int out_size, void* d_ws, size_t ws_size,
                              hipStream_t stream) {
  const float* feat    = (const float*)d_in[0];
  const int*   enc_src = (const int*)d_in[1];
  const int*   enc_dst = (const int*)d_in[2];
  const int*   dec_src = (const int*)d_in[3];
  const int*   dec_dst = (const int*)d_in[4];
  const float* W_enc   = (const float*)d_in[5];
  const float* b_enc   = (const float*)d_in[6];
  const float* W_proc  = (const float*)d_in[7];
  const float* b_proc  = (const float*)d_in[8];
  const float* W_dec   = (const float*)d_in[9];
  const float* b_dec   = (const float*)d_in[10];
  const float* W_lin   = (const float*)d_in[11];
  const float* b_lin   = (const float*)d_in[12];
  const float* W_a1    = (const float*)d_in[13];
  const float* b_a1    = (const float*)d_in[14];
  const float* W_a2    = (const float*)d_in[15];
  const float* b_a2    = (const float*)d_in[16];
  const float* W_o     = (const float*)d_in[17];
  const float* b_o     = (const float*)d_in[18];
  float* out = (float*)d_out;

  // workspace layout
  float* c_se = (float*)d_ws;          // N_F
  float* c_de = c_se + N_F;            // N_H
  float* c_sd = c_de + N_H;            // N_H
  float* c_dd = c_sd + N_H;            // N_F
  float* agg  = c_dd + N_F;            // N_H*12
  float* y0   = agg + (size_t)N_H * T_IN;      // N_F
  float* mbuf = y0 + N_F;              // N_H*64
  float* abuf = mbuf + (size_t)N_H * HDIM;     // N_F*64

  const int B = 256;

  // GCN norms
  zero_kernel<<<cdiv(N_F, B), B, 0, stream>>>(c_se, N_F);
  zero_kernel<<<cdiv(N_H, B), B, 0, stream>>>(c_de, N_H);
  zero_kernel<<<cdiv(N_H, B), B, 0, stream>>>(c_sd, N_H);
  zero_kernel<<<cdiv(N_F, B), B, 0, stream>>>(c_dd, N_F);
  zero_kernel<<<cdiv((long long)N_H * T_IN, B), B, 0, stream>>>(agg, N_H * T_IN);
  deg_kernel<<<cdiv(NEDGE, B), B, 0, stream>>>(enc_src, enc_dst, c_se, c_de, NEDGE);
  deg_kernel<<<cdiv(NEDGE, B), B, 0, stream>>>(dec_src, dec_dst, c_sd, c_dd, NEDGE);
  invsqrt_kernel<<<cdiv(N_F, B), B, 0, stream>>>(c_se, N_F);
  invsqrt_kernel<<<cdiv(N_H, B), B, 0, stream>>>(c_de, N_H);
  invsqrt_kernel<<<cdiv(N_H, B), B, 0, stream>>>(c_sd, N_H);
  invsqrt_kernel<<<cdiv(N_F, B), B, 0, stream>>>(c_dd, N_F);

  // encoder aggregation + y0
  enc_agg_kernel<<<cdiv((long long)NEDGE * T_IN, B), B, 0, stream>>>(
      feat, enc_src, enc_dst, c_se, agg, NEDGE);
  y0_kernel<<<cdiv(N_F, B), B, 0, stream>>>(feat, y0, N_F);

  // decoder scan
  const float* yprev = y0;
  for (int o = 0; o < T_OUT; ++o) {
    zero_kernel<<<cdiv((long long)N_F * HDIM, B), B, 0, stream>>>(abuf, N_F * HDIM);
    compute_m_kernel<<<cdiv((long long)N_H * HDIM, B), B, 0, stream>>>(
        agg, c_de, c_sd, W_enc, b_enc, W_proc, b_proc, mbuf, o, N_H);
    dec_agg_kernel<<<cdiv((long long)NEDGE * 16, B), B, 0, stream>>>(
        mbuf, dec_src, dec_dst, abuf, NEDGE);
    decoder_step_kernel<<<cdiv(N_F, 64), 128, 0, stream>>>(
        abuf, c_dd, yprev,
        W_dec + (size_t)o * HDIM * HDIM, b_dec + (size_t)o * HDIM,
        W_lin, b_lin, W_a1, b_a1, W_a2, b_a2, W_o, b_o,
        out + (size_t)o * N_F, N_F);
    yprev = out + (size_t)o * N_F;
  }
}
